// GCNClassifier_77910706750016
// MI455X (gfx1250) — matrix operand order repaired and checked by reading, not compiled, not run on hardware
//
#include <hip/hip_runtime.h>
#include <math.h>

#define D 128
#define EPS 1e-5f

typedef __attribute__((ext_vector_type(16))) __bf16 v16bf;
typedef __attribute__((ext_vector_type(8)))  float  v8f;

union Frag {
  v16bf v;
  __bf16 e[16];
  uint4  q[2];
};

__device__ __forceinline__ void cvt8(const float* __restrict__ p, __bf16* d) {
  const float4 a = ((const float4*)p)[0];
  const float4 b = ((const float4*)p)[1];
  d[0] = (__bf16)a.x; d[1] = (__bf16)a.y; d[2] = (__bf16)a.z; d[3] = (__bf16)a.w;
  d[4] = (__bf16)b.x; d[5] = (__bf16)b.y; d[6] = (__bf16)b.z; d[7] = (__bf16)b.w;
}

// ---------------------------------------------------------------------------
// Edge scatter: agg[dst] += w * feat[src].  One wave32 per edge; lane l moves
// channels [4l,4l+4) -> one fully-coalesced 512B row read per wave, then 4
// global_atomic_add_f32 per lane.  Working set (51MB features) fits in L2.
// ---------------------------------------------------------------------------
__global__ __launch_bounds__(256) void scatter_edges(
    const float* __restrict__ feat, const int* __restrict__ src,
    const int* __restrict__ dst, const float* __restrict__ ew,
    float* __restrict__ agg, int nEdges) {
  const int wave = (int)((blockIdx.x * blockDim.x + threadIdx.x) >> 5);
  const int lane = threadIdx.x & 31;
  if (wave >= nEdges) return;
  const int   s = src[wave];
  const int   t = dst[wave];
  const float w = ew[wave];
  const float4 v = ((const float4*)(feat + (size_t)s * D))[lane];
  float* o = agg + (size_t)t * D + lane * 4;
  atomicAdd(o + 0, w * v.x);
  atomicAdd(o + 1, w * v.y);
  atomicAdd(o + 2, w * v.z);
  atomicAdd(o + 3, w * v.w);
}

// ---------------------------------------------------------------------------
// Fused GEMM + bias + BN(eval) + ReLU:
//   out = relu(bn(A @ Wrel + (HAS_ROOT ? X @ Wroot : 0) + bias))
// Block = 256 threads = 8 waves; block tile 128 rows x 128 cols.
// Wave w owns rows [16w,16w+16) and all 8 16-col tiles (8 v8f accumulators).
// K loop: 4 steps of v_wmma_f32_16x16x32_bf16 (x2 when fusing the root GEMM).
// Weights staged in LDS as bf16, transposed to [n][k] with an XOR swizzle on
// 8-element k-chunks so B-fragment ds_load_b128 reads are bank-conflict free.
// ---------------------------------------------------------------------------
template <bool HAS_ROOT>
__global__ __launch_bounds__(256) void gemm_bn_relu(
    const float* __restrict__ A, const float* __restrict__ X,
    const float* __restrict__ Wrel, const float* __restrict__ Wroot,
    const float* __restrict__ bias,
    const float* __restrict__ gam, const float* __restrict__ bet,
    const float* __restrict__ rmean, const float* __restrict__ rvar,
    float* __restrict__ out, int nrows) {
  __shared__ __align__(16) __bf16 sW[(HAS_ROOT ? 2 : 1) * D * D];

  for (int i = threadIdx.x; i < D * D; i += 256) {
    const int k = i >> 7, n = i & 127;                       // W is [k][n] row-major
    const int sw = (n << 7) | (((k >> 3) ^ (n & 15)) << 3) | (k & 7);
    sW[sw] = (__bf16)Wrel[i];
    if (HAS_ROOT) sW[D * D + sw] = (__bf16)Wroot[i];
  }
  __syncthreads();

  const int wave = threadIdx.x >> 5;
  const int lane = threadIdx.x & 31;
  const int half = lane >> 4;       // which 16-lane half
  const int lm   = lane & 15;
  const int rowBase = blockIdx.x * 128 + wave * 16;
  const int arow    = rowBase + lm;
  const bool rowOK  = arow < nrows;

  v8f acc[8] = {};

  for (int k0 = 0; k0 < D; k0 += 32) {
    // A fragment, 16-bit 16x32 layout (ISA 7.12.2): lane half h holds
    // K = k0+8h+0..7 in e[0..7] and K = k0+16+8h+0..7 in e[8..15].
    Frag a, xf;
    if (rowOK) {
      cvt8(A + (size_t)arow * D + k0 + half * 8,      &a.e[0]);
      cvt8(A + (size_t)arow * D + k0 + 16 + half * 8, &a.e[8]);
      if (HAS_ROOT) {
        cvt8(X + (size_t)arow * D + k0 + half * 8,      &xf.e[0]);
        cvt8(X + (size_t)arow * D + k0 + 16 + half * 8, &xf.e[8]);
      }
    } else {
      #pragma unroll
      for (int i = 0; i < 16; ++i) {
        a.e[i] = (__bf16)0.f;
        if (HAS_ROOT) xf.e[i] = (__bf16)0.f;
      }
    }

    // B fragment: lane = column n, half h holds K = k0+8h+0..7 and k0+16+8h+0..7.
    const int kk = k0 + half * 8;
    const int c0 = (((kk >> 3) + 0) ^ lm) << 3;   // swizzled chunk offsets
    const int c1 = (((kk >> 3) + 2) ^ lm) << 3;

    #pragma unroll
    for (int nt = 0; nt < 8; ++nt) {
      const int n = nt * 16 + lm;
      Frag b;
      b.q[0] = *(const uint4*)&sW[(n << 7) + c0];
      b.q[1] = *(const uint4*)&sW[(n << 7) + c1];
      acc[nt] = __builtin_amdgcn_wmma_f32_16x16x32_bf16(
          false, a.v, false, b.v, (short)0, acc[nt], false, false);
      if (HAS_ROOT) {
        Frag b2;
        b2.q[0] = *(const uint4*)&sW[D * D + (n << 7) + c0];
        b2.q[1] = *(const uint4*)&sW[D * D + (n << 7) + c1];
        acc[nt] = __builtin_amdgcn_wmma_f32_16x16x32_bf16(
            false, xf.v, false, b2.v, (short)0, acc[nt], false, false);
      }
    }
  }

  // Epilogue: C/D layout — lane = col, VGPR r = row (r + 8*half).
  #pragma unroll
  for (int nt = 0; nt < 8; ++nt) {
    const int ocol = nt * 16 + lm;
    const float sc  = gam[ocol] * rsqrtf(rvar[ocol] + EPS);
    const float ofs = bet[ocol] - rmean[ocol] * sc;
    const float bs  = bias[ocol];
    #pragma unroll
    for (int r = 0; r < 8; ++r) {
      const int orow = rowBase + half * 8 + r;
      if (orow < nrows) {
        const float h = (acc[nt][r] + bs) * sc + ofs;
        out[(size_t)orow * D + ocol] = fmaxf(h, 0.f);
      }
    }
  }
}

// ---------------------------------------------------------------------------
// Output head: sigmoid(h @ Wout + bout).  One wave32 per row; float4 per lane
// then a 5-step shuffle reduction.
// ---------------------------------------------------------------------------
__global__ __launch_bounds__(256) void head_sigmoid(
    const float* __restrict__ h, const float* __restrict__ Wout,
    const float* __restrict__ bout, float* __restrict__ out, int nrows) {
  const int wave = (int)((blockIdx.x * blockDim.x + threadIdx.x) >> 5);
  const int lane = threadIdx.x & 31;
  if (wave >= nrows) return;
  const float4 a = ((const float4*)(h + (size_t)wave * D))[lane];
  const float4 b = ((const float4*)Wout)[lane];
  float s = a.x * b.x + a.y * b.y + a.z * b.z + a.w * b.w;
  #pragma unroll
  for (int off = 16; off > 0; off >>= 1) s += __shfl_down(s, off, 32);
  if (lane == 0) out[wave] = 1.0f / (1.0f + __expf(-(s + bout[0])));
}

// ---------------------------------------------------------------------------
extern "C" void kernel_launch(void* const* d_in, const int* in_sizes, int n_in,
                              void* d_out, int out_size, void* d_ws, size_t ws_size,
                              hipStream_t stream) {
  const float* x     = (const float*)d_in[0];
  const int*   ei    = (const int*)  d_in[1];
  const float* ew    = (const float*)d_in[2];
  const float* Wrel0 = (const float*)d_in[3];
  const float* brel0 = (const float*)d_in[4];
  const float* Wroot0= (const float*)d_in[5];
  const float* g0    = (const float*)d_in[6];
  const float* be0   = (const float*)d_in[7];
  const float* rm0   = (const float*)d_in[8];
  const float* rv0   = (const float*)d_in[9];
  const float* Wrel1 = (const float*)d_in[10];
  const float* brel1 = (const float*)d_in[11];
  const float* Wroot1= (const float*)d_in[12];
  const float* g1    = (const float*)d_in[13];
  const float* be1   = (const float*)d_in[14];
  const float* rm1   = (const float*)d_in[15];
  const float* rv1   = (const float*)d_in[16];
  const float* Wm0   = (const float*)d_in[17];
  const float* bm0   = (const float*)d_in[18];
  const float* gm0   = (const float*)d_in[19];
  const float* bem0  = (const float*)d_in[20];
  const float* rmm0  = (const float*)d_in[21];
  const float* rvm0  = (const float*)d_in[22];
  const float* Wout  = (const float*)d_in[23];
  const float* bout  = (const float*)d_in[24];

  const int N = in_sizes[0] / D;
  const int E = in_sizes[2];
  const int* src = ei;
  const int* dst = ei + E;

  const size_t ND = (size_t)N * D;
  float* buf0 = (float*)d_ws;      // agg0 -> h1
  float* buf1 = buf0 + ND;         // h0
  float* buf2 = buf1 + ND;         // agg1 -> h2

  const dim3 blk(256);
  const dim3 gScat((E + 7) / 8);
  const dim3 gGemm((N + 127) / 128);
  const dim3 gHead((N + 7) / 8);

  // Layer 0: conv -> BN -> ReLU
  hipMemsetAsync(buf0, 0, ND * sizeof(float), stream);
  scatter_edges<<<gScat, blk, 0, stream>>>(x, src, dst, ew, buf0, E);
  gemm_bn_relu<true><<<gGemm, blk, 0, stream>>>(buf0, x, Wrel0, Wroot0, brel0,
                                                g0, be0, rm0, rv0, buf1, N);
  // Layer 1
  hipMemsetAsync(buf2, 0, ND * sizeof(float), stream);
  scatter_edges<<<gScat, blk, 0, stream>>>(buf1, src, dst, ew, buf2, E);
  gemm_bn_relu<true><<<gGemm, blk, 0, stream>>>(buf2, buf1, Wrel1, Wroot1, brel1,
                                                g1, be1, rm1, rv1, buf0, N);
  // MLP layer (no root term)
  gemm_bn_relu<false><<<gGemm, blk, 0, stream>>>(buf0, nullptr, Wm0, nullptr, bm0,
                                                 gm0, bem0, rmm0, rvm0, buf2, N);
  // Output head
  head_sigmoid<<<gHead, blk, 0, stream>>>(buf2, Wout, bout, (float*)d_out, N);
}
